// Attention_77094662963280
// MI455X (gfx1250) — compile-verified, matrix operand order checked by reading, not run on hardware
//
#include <hip/hip_runtime.h>
#include <stdint.h>

typedef __attribute__((ext_vector_type(16))) __bf16    v16bf;
typedef __attribute__((ext_vector_type(8)))  float     v8f;
typedef __attribute__((ext_vector_type(8)))  uint16_t  v8u16;
typedef __attribute__((ext_vector_type(16))) uint16_t  v16u16;
typedef __attribute__((ext_vector_type(4)))  unsigned int v4u;
typedef __attribute__((ext_vector_type(8)))  unsigned int v8u;
typedef __attribute__((ext_vector_type(8)))  int       v8i;
typedef __attribute__((ext_vector_type(4)))  int       v4i;

#if defined(__has_builtin)
#if __has_builtin(__builtin_amdgcn_tensor_load_to_lds) && __has_builtin(__builtin_amdgcn_s_wait_tensorcnt)
#define USE_TDM 1
#endif
#endif
#ifndef USE_TDM
#define USE_TDM 0
#endif

__device__ __forceinline__ uint16_t f32_to_bf16_bits(float f) {
  uint32_t x = __builtin_bit_cast(uint32_t, f);
  x += 0x7FFFu + ((x >> 16) & 1u);          // round-to-nearest-even
  return (uint16_t)(x >> 16);
}
__device__ __forceinline__ float bf16_lo(uint32_t u) { return __builtin_bit_cast(float, u << 16); }
__device__ __forceinline__ float bf16_hi(uint32_t u) { return __builtin_bit_cast(float, u & 0xFFFF0000u); }
__device__ __forceinline__ uint32_t pack_bf16x2(float lo, float hi) {
  return (uint32_t)f32_to_bf16_bits(lo) | ((uint32_t)f32_to_bf16_bits(hi) << 16);
}

// ---------------------------------------------------------------- weights cvt
__global__ void cvt_kernel(const float* __restrict__ s, uint16_t* __restrict__ d, int n) {
  int i = blockIdx.x * 256 + threadIdx.x;
  if (i < n) d[i] = f32_to_bf16_bits(s[i]);
}

// ---------------------------------------------------------------- WMMA GEMM
constexpr int KDIM = 192;
constexpr int BM   = 64;
constexpr int BN   = 96;
constexpr int LDK  = 200;   // padded LDS row stride; 400B, 16B-aligned, conflict-free

__device__ __forceinline__ v16bf load_frag(const uint16_t* rowp, int kbase) {
  v8u16 lo = *(const v8u16*)(rowp + kbase);
  v8u16 hi = *(const v8u16*)(rowp + kbase + 16);
  v16u16 u;
#pragma unroll
  for (int i = 0; i < 8; ++i) { u[i] = lo[i]; u[i + 8] = hi[i]; }
  return __builtin_bit_cast(v16bf, u);
}

template <bool A_F32, bool OUT_BF16>
__global__ __launch_bounds__(256) void gemm_wmma(const void* __restrict__ Av,
                                                 const uint16_t* __restrict__ B,
                                                 void* __restrict__ Dv, int Ntot) {
  __shared__ __attribute__((aligned(16))) uint16_t As[BM * LDK];
  __shared__ __attribute__((aligned(16))) uint16_t Bs[BN * LDK];

  const int tid = threadIdx.x;
  const size_t m0 = (size_t)blockIdx.y * BM;
  const int n0 = blockIdx.x * BN;

  if constexpr (A_F32) {
    const float* A = (const float*)Av;
#pragma unroll
    for (int j = 0; j < 12; ++j) {
      int idx = tid + 256 * j;
      int row = idx / 48, c4 = idx % 48;
      float4 f = *(const float4*)(A + (m0 + row) * KDIM + c4 * 4);
      uint16_t* d = &As[row * LDK + c4 * 4];
      d[0] = f32_to_bf16_bits(f.x);
      d[1] = f32_to_bf16_bits(f.y);
      d[2] = f32_to_bf16_bits(f.z);
      d[3] = f32_to_bf16_bits(f.w);
    }
  } else {
    const uint16_t* A = (const uint16_t*)Av;
#pragma unroll
    for (int j = 0; j < 12; ++j) {
      int idx = tid + 256 * j;
      int row = idx / 48, c4 = idx % 48;
      *(uint2*)&As[row * LDK + c4 * 4] = *(const uint2*)(A + (m0 + row) * KDIM + c4 * 4);
    }
  }
#pragma unroll
  for (int j = 0; j < 18; ++j) {
    int idx = tid + 256 * j;
    int row = idx / 48, c4 = idx % 48;
    *(uint2*)&Bs[row * LDK + c4 * 4] = *(const uint2*)(B + (size_t)(n0 + row) * KDIM + c4 * 4);
  }
  __syncthreads();

  const int lane = tid & 31;
  const int wave = tid >> 5;
  const int wrow = wave & 3;
  const int wcol = wave >> 2;
  const int half = lane >> 4;
  const int l16 = lane & 15;

  const uint16_t* arow  = &As[(wrow * 16 + l16) * LDK];
  const uint16_t* brow0 = &Bs[(wcol * 48 +  0 + l16) * LDK];
  const uint16_t* brow1 = &Bs[(wcol * 48 + 16 + l16) * LDK];
  const uint16_t* brow2 = &Bs[(wcol * 48 + 32 + l16) * LDK];

  v8f acc[3] = {};
#pragma unroll
  for (int kc = 0; kc < KDIM; kc += 32) {
    const int kbase = kc + 8 * half;
    v16bf a  = load_frag(arow,  kbase);
    v16bf b0 = load_frag(brow0, kbase);
    v16bf b1 = load_frag(brow1, kbase);
    v16bf b2 = load_frag(brow2, kbase);
    acc[0] = __builtin_amdgcn_wmma_f32_16x16x32_bf16(false, a, false, b0, (short)0, acc[0], false, false);
    acc[1] = __builtin_amdgcn_wmma_f32_16x16x32_bf16(false, a, false, b1, (short)0, acc[1], false, false);
    acc[2] = __builtin_amdgcn_wmma_f32_16x16x32_bf16(false, a, false, b2, (short)0, acc[2], false, false);
  }

  const size_t rbase = (m0 + (size_t)(wrow * 16 + 8 * half)) * (size_t)Ntot;
#pragma unroll
  for (int f = 0; f < 3; ++f) {
    const int col = n0 + wcol * 48 + f * 16 + l16;
#pragma unroll
    for (int i = 0; i < 8; ++i) {
      size_t o = rbase + (size_t)i * Ntot + col;
      if constexpr (OUT_BF16) ((uint16_t*)Dv)[o] = f32_to_bf16_bits(acc[f][i]);
      else                    ((float*)Dv)[o]    = acc[f][i];
    }
  }
}

// ---------------------------------------------------------------- TDM helper
#if USE_TDM
// 2-D tensor tile gather into LDS via the Tensor Data Mover (D# per ISA ch.8).
// tile/tensor dims in elements (data_size = 2 bytes), stride0 in elements.
// amdgpu-toolchain (clang-23) 6-arg builtin: (v4u, v8i, v4i, v4i, v8i, i32 cpol).
__device__ __forceinline__ void tdm_load_2d(uint32_t lds_addr, const void* gptr,
                                            uint32_t d0, uint32_t d1, uint32_t stride0) {
  uint64_t ga = (uint64_t)(uintptr_t)gptr;
  v4u g0;
  g0[0] = 1u;                                           // count=1, user descriptor
  g0[1] = lds_addr;                                     // LDS byte address
  g0[2] = (uint32_t)ga;                                 // global_addr[31:0]
  g0[3] = (uint32_t)((ga >> 32) & 0x01FFFFFFu) | (2u << 30);  // addr[56:32] | type=2
  v8u g1;
  g1[0] = 1u << 16;                                     // data_size = 1 (2 bytes)
  g1[1] = (d0 & 0xFFFFu) << 16;                         // tensor_dim0[15:0]
  g1[2] = ((d0 >> 16) & 0xFFFFu) | ((d1 & 0xFFFFu) << 16);     // dim0 hi | dim1 lo
  g1[3] = ((d1 >> 16) & 0xFFFFu) | ((d0 & 0xFFFFu) << 16);     // dim1 hi | tile_dim0
  g1[4] = d1 & 0xFFFFu;                                 // tile_dim1 (tile_dim2 = 0)
  g1[5] = stride0;                                      // tensor_dim0_stride[31:0]
  g1[6] = 0u;
  g1[7] = 0u;
  v4i z4 = {0, 0, 0, 0};
  v8i z8 = {0, 0, 0, 0, 0, 0, 0, 0};
  __builtin_amdgcn_tensor_load_to_lds(g0, __builtin_bit_cast(v8i, g1), z4, z4, z8, 0);
}
__device__ __forceinline__ uint32_t lds_off(const void* p) {
  return (uint32_t)(uintptr_t)p;   // LDS generic addr: low 32 bits = LDS offset
}
#endif

// ---------------------------------------------------------------- attention mid
// One workgroup per (b, hd). score[m,l] = sum_n q[n,m]*k[n,l]; softmax rows; out = W*v.
__global__ __launch_bounds__(256) void attn_mid(const uint16_t* __restrict__ qkv,
                                                uint16_t* __restrict__ tmp) {
  constexpr int NPIX = 16384, C3 = 576, C = 192;
  const int b = blockIdx.x >> 5;
  const int hd = blockIdx.x & 31;
  const size_t row0 = (size_t)b * NPIX;
  const int qo = hd * 6, ko = C + hd * 6, vo = 2 * C + hd * 6;

  float S[6][6];
#pragma unroll
  for (int m = 0; m < 6; ++m)
#pragma unroll
    for (int l = 0; l < 6; ++l) S[m][l] = 0.f;

#if USE_TDM
  constexpr int CH = 1024;                       // rows per TDM chunk (16 chunks)
  constexpr int NCHUNK = NPIX / CH;
  __shared__ __attribute__((aligned(16))) uint16_t qb[2][CH * 6];
  __shared__ __attribute__((aligned(16))) uint16_t kb[2][CH * 6];
  const uint16_t* qbase = qkv + (row0 * C3 + qo);
  const uint16_t* kbase = qkv + (row0 * C3 + ko);
  const uint16_t* vbase = qkv + (row0 * C3 + vo);
  const bool issuer = (threadIdx.x < 32);        // wave 0 (uniform per wave)

  if (issuer) {                                  // prime the pipeline: chunk 0
    tdm_load_2d(lds_off(&qb[0][0]), qbase, 6, CH, C3);
    tdm_load_2d(lds_off(&kb[0][0]), kbase, 6, CH, C3);
  }
  for (int c = 0; c < NCHUNK; ++c) {
    const int p = c & 1;
    __syncthreads();                             // buffer p^1 free for prefetch
    if (issuer) {
      if (c + 1 < NCHUNK) {
        tdm_load_2d(lds_off(&qb[p ^ 1][0]), qbase + (size_t)(c + 1) * CH * C3, 6, CH, C3);
        tdm_load_2d(lds_off(&kb[p ^ 1][0]), kbase + (size_t)(c + 1) * CH * C3, 6, CH, C3);
        __builtin_amdgcn_s_wait_tensorcnt(2);    // in-order: chunk c complete
      } else {
        __builtin_amdgcn_s_wait_tensorcnt(0);
      }
    }
    __syncthreads();                             // chunk c visible to all waves
    for (int r = threadIdx.x; r < CH; r += 256) {
      const uint32_t* qp = (const uint32_t*)&qb[p][r * 6];
      const uint32_t* kp = (const uint32_t*)&kb[p][r * 6];
      uint32_t q0 = qp[0], q1 = qp[1], q2 = qp[2];
      uint32_t k0 = kp[0], k1 = kp[1], k2 = kp[2];
      float q[6] = {bf16_lo(q0), bf16_hi(q0), bf16_lo(q1), bf16_hi(q1), bf16_lo(q2), bf16_hi(q2)};
      float k[6] = {bf16_lo(k0), bf16_hi(k0), bf16_lo(k1), bf16_hi(k1), bf16_lo(k2), bf16_hi(k2)};
#pragma unroll
      for (int m = 0; m < 6; ++m)
#pragma unroll
        for (int l = 0; l < 6; ++l) S[m][l] = fmaf(q[m], k[l], S[m][l]);
    }
  }
#else
  for (int n = threadIdx.x; n < NPIX; n += 256) {
    const uint32_t* p = (const uint32_t*)(qkv + (row0 + n) * C3);
    uint32_t q0 = p[qo / 2], q1 = p[qo / 2 + 1], q2 = p[qo / 2 + 2];
    uint32_t k0 = p[ko / 2], k1 = p[ko / 2 + 1], k2 = p[ko / 2 + 2];
    float q[6] = {bf16_lo(q0), bf16_hi(q0), bf16_lo(q1), bf16_hi(q1), bf16_lo(q2), bf16_hi(q2)};
    float k[6] = {bf16_lo(k0), bf16_hi(k0), bf16_lo(k1), bf16_hi(k1), bf16_lo(k2), bf16_hi(k2)};
#pragma unroll
    for (int m = 0; m < 6; ++m)
#pragma unroll
      for (int l = 0; l < 6; ++l) S[m][l] = fmaf(q[m], k[l], S[m][l]);
  }
#endif

  // wave32 reduction, then cross-wave combine in LDS
#pragma unroll
  for (int m = 0; m < 6; ++m)
#pragma unroll
    for (int l = 0; l < 6; ++l) {
      float v = S[m][l];
      for (int off = 16; off; off >>= 1) v += __shfl_down(v, off, 32);
      S[m][l] = v;
    }
  __shared__ float Ssum[36];
  __shared__ float Wt[36];
  if (threadIdx.x < 36) Ssum[threadIdx.x] = 0.f;
  __syncthreads();
  if ((threadIdx.x & 31) == 0) {
#pragma unroll
    for (int m = 0; m < 6; ++m)
#pragma unroll
      for (int l = 0; l < 6; ++l) atomicAdd(&Ssum[m * 6 + l], S[m][l]);
  }
  __syncthreads();
  if (threadIdx.x < 6) {
    const float TEMP = 0.17677669529663689f;     // 32^-0.5
    const int m = threadIdx.x;
    float s[6], mx = -1e30f;
#pragma unroll
    for (int l = 0; l < 6; ++l) { s[l] = Ssum[m * 6 + l] * TEMP; mx = fmaxf(mx, s[l]); }
    float sum = 0.f;
#pragma unroll
    for (int l = 0; l < 6; ++l) { s[l] = __expf(s[l] - mx); sum += s[l]; }
    float inv = 1.f / sum;
#pragma unroll
    for (int l = 0; l < 6; ++l) Wt[m * 6 + l] = s[l] * inv;
  }
  __syncthreads();

  float W[6][6];
#pragma unroll
  for (int m = 0; m < 6; ++m)
#pragma unroll
    for (int l = 0; l < 6; ++l) W[m][l] = Wt[m * 6 + l];

#if USE_TDM
  if (issuer)                                    // prime v chunk 0 (reuse qb)
    tdm_load_2d(lds_off(&qb[0][0]), vbase, 6, CH, C3);
  for (int c = 0; c < NCHUNK; ++c) {
    const int p = c & 1;
    __syncthreads();
    if (issuer) {
      if (c + 1 < NCHUNK) {
        tdm_load_2d(lds_off(&qb[p ^ 1][0]), vbase + (size_t)(c + 1) * CH * C3, 6, CH, C3);
        __builtin_amdgcn_s_wait_tensorcnt(1);
      } else {
        __builtin_amdgcn_s_wait_tensorcnt(0);
      }
    }
    __syncthreads();
    for (int r = threadIdx.x; r < CH; r += 256) {
      const uint32_t* vp = (const uint32_t*)&qb[p][r * 6];
      uint32_t v0 = vp[0], v1 = vp[1], v2 = vp[2];
      float v[6] = {bf16_lo(v0), bf16_hi(v0), bf16_lo(v1), bf16_hi(v1), bf16_lo(v2), bf16_hi(v2)};
      float o[6];
#pragma unroll
      for (int m = 0; m < 6; ++m) {
        float a = 0.f;
#pragma unroll
        for (int l = 0; l < 6; ++l) a = fmaf(W[m][l], v[l], a);
        o[m] = a;
      }
      uint32_t* d = (uint32_t*)(tmp + (row0 + c * CH + r) * C + qo);
      d[0] = pack_bf16x2(o[0], o[1]);
      d[1] = pack_bf16x2(o[2], o[3]);
      d[2] = pack_bf16x2(o[4], o[5]);
    }
  }
#else
  for (int n = threadIdx.x; n < NPIX; n += 256) {
    const uint32_t* p = (const uint32_t*)(qkv + (row0 + n) * C3);
    uint32_t v0 = p[vo / 2], v1 = p[vo / 2 + 1], v2 = p[vo / 2 + 2];
    float v[6] = {bf16_lo(v0), bf16_hi(v0), bf16_lo(v1), bf16_hi(v1), bf16_lo(v2), bf16_hi(v2)};
    float o[6];
#pragma unroll
    for (int m = 0; m < 6; ++m) {
      float a = 0.f;
#pragma unroll
      for (int l = 0; l < 6; ++l) a = fmaf(W[m][l], v[l], a);
      o[m] = a;
    }
    uint32_t* d = (uint32_t*)(tmp + (row0 + n) * C + qo);
    d[0] = pack_bf16x2(o[0], o[1]);
    d[1] = pack_bf16x2(o[2], o[3]);
    d[2] = pack_bf16x2(o[4], o[5]);
  }
#endif
}

// ---------------------------------------------------------------- launch
extern "C" void kernel_launch(void* const* d_in, const int* in_sizes, int n_in,
                              void* d_out, int out_size, void* d_ws, size_t ws_size,
                              hipStream_t stream) {
  (void)in_sizes; (void)n_in; (void)out_size; (void)ws_size;
  const float* x      = (const float*)d_in[0];   // [262144,192]
  const float* w_qkv  = (const float*)d_in[1];   // [576,192]
  const float* w_proj = (const float*)d_in[2];   // [192,192]
  float* out = (float*)d_out;                    // [262144,192]

  constexpr size_t MROWS = 262144;               // B*H*W
  uint8_t* ws = (uint8_t*)d_ws;
  size_t off = 0;
  uint16_t* qkv = (uint16_t*)(ws + off); off += MROWS * 576 * 2;   // 288 MB
  uint16_t* tmp = (uint16_t*)(ws + off); off += MROWS * 192 * 2;   //  96 MB
  uint16_t* wq  = (uint16_t*)(ws + off); off += (size_t)576 * 192 * 2;
  uint16_t* wp  = (uint16_t*)(ws + off); off += (size_t)192 * 192 * 2;

  cvt_kernel<<<(110592 + 255) / 256, 256, 0, stream>>>(w_qkv, wq, 110592);
  cvt_kernel<<<(36864 + 255) / 256, 256, 0, stream>>>(w_proj, wp, 36864);

  // QKV projection: x[262144,192] * wq[576,192]^T -> qkv (bf16)
  gemm_wmma<true, true><<<dim3(576 / BN, MROWS / BM), 256, 0, stream>>>(x, wq, qkv, 576);

  // per-(b,hd) head attention + softmax -> tmp (bf16, channel = hd*6+m)
  attn_mid<<<512, 256, 0, stream>>>(qkv, tmp);

  // output projection: tmp[262144,192] * wp[192,192]^T -> out (fp32)
  gemm_wmma<false, false><<<dim3(192 / BN, MROWS / BM), 256, 0, stream>>>(tmp, wp, out, 192);
}
